// CausalAttention_26603027431614
// MI455X (gfx1250) — matrix operand order repaired and checked by reading, not compiled, not run on hardware
//
#include <hip/hip_runtime.h>

// Causal attention, B=4, S=4096, D=768, f32 in/out.
// Prep : Wq/Wk/Wv f32[k][n] -> f16 transposed Wt[n][k] (L2-resident, 3.5MB).
// Phase1: Q,K (row-major f16) and V (transposed f16 Vt[d][s]) via WMMA.
// Phase2: flash attention, 128-key chunks, WMMA QK^T and PV, async Q staging.

typedef __attribute__((ext_vector_type(16))) _Float16 v16h;
typedef __attribute__((ext_vector_type(8)))  _Float16 v8h;
typedef __attribute__((ext_vector_type(8)))  float    v8f;
typedef __attribute__((ext_vector_type(4)))  float    v4f;

#define DIM    768
#define SEQ    4096
#define BATCH  4
#define KC     128   // key chunk
#define QT     16    // query tile

static __device__ __forceinline__ v8f wmma_f16(v16h a, v16h b, v8f c) {
  return __builtin_amdgcn_wmma_f32_16x16x32_f16(
      /*neg_a=*/false, a, /*neg_b=*/false, b,
      /*c_mod=*/(short)0, c, /*reuse_a=*/false, /*reuse_b=*/false);
}

// ---- A operand (16x32 f16), source row-major, 16B-aligned vector loads.
//      Lane<16: row M=lane, K = k0+0..7 & k0+16..23; lane>=16: +8 shifted.
static __device__ __forceinline__ v16h load_A_f16(const _Float16* base, int ld,
                                                  int row0, int k0, int lane) {
  const int m    = lane & 15;
  const int koff = k0 + ((lane >> 4) << 3);
  const _Float16* p = base + (size_t)(row0 + m) * ld + koff;
  v8h lo = *(const v8h*)p;
  v8h hi = *(const v8h*)(p + 16);
  return __builtin_shufflevector(lo, hi, 0, 1, 2, 3, 4, 5, 6, 7,
                                 8, 9, 10, 11, 12, 13, 14, 15);
}

// ---- A operand from f32 source (x), wide loads then pack-convert.
static __device__ __forceinline__ v16h load_A_f32cvt(const float* base, int ld,
                                                     int row0, int k0, int lane) {
  const int m    = lane & 15;
  const int koff = k0 + ((lane >> 4) << 3);
  const float* p = base + (size_t)(row0 + m) * ld + koff;
  v4f x0 = *(const v4f*)(p);
  v4f x1 = *(const v4f*)(p + 4);
  v4f x2 = *(const v4f*)(p + 16);
  v4f x3 = *(const v4f*)(p + 20);
  v16h a;
#pragma unroll
  for (int i = 0; i < 4; ++i) {
    a[i]      = (_Float16)x0[i];
    a[i + 4]  = (_Float16)x1[i];
    a[i + 8]  = (_Float16)x2[i];
    a[i + 12] = (_Float16)x3[i];
  }
  return a;
}

// ---- B operand (32x16) from a TRANSPOSED source: B[kk][n] = T[n0+n][k0+kk].
//      Each lane reads K = k0+8h+0..7 & k0+16+8h+0..7 (two b128, 16B-aligned).
static __device__ __forceinline__ v16h load_BT_f16(const _Float16* base, int ld,
                                                   int n0, int k0, int lane) {
  const _Float16* p =
      base + (size_t)(n0 + (lane & 15)) * ld + k0 + ((lane >> 4) << 3);
  v8h lo = *(const v8h*)p;
  v8h hi = *(const v8h*)(p + 16);
  return __builtin_shufflevector(lo, hi, 0, 1, 2, 3, 4, 5, 6, 7,
                                 8, 9, 10, 11, 12, 13, 14, 15);
}

// ---- C/D (16x16 f32): lane<16 col=lane rows 0..7; lane>=16 col=lane-16 rows 8..15.
static __device__ __forceinline__ void store_C_f16(_Float16* base, int ld,
                                                   int row0, int col0, int lane,
                                                   v8f c) {
  const int n  = col0 + (lane & 15);
  const int mb = row0 + ((lane >> 4) << 3);
#pragma unroll
  for (int r = 0; r < 8; ++r) base[(size_t)(mb + r) * ld + n] = (_Float16)c[r];
}

// ---- C store into a transposed f16 matrix T[col][row]: contiguous 8 halfs.
static __device__ __forceinline__ void store_C_f16_T(_Float16* baseT, int ldT,
                                                     int row0, int col0,
                                                     int lane, v8f c) {
  const int d  = col0 + (lane & 15);
  const int sb = row0 + ((lane >> 4) << 3);
  v8h h;
#pragma unroll
  for (int r = 0; r < 8; ++r) h[r] = (_Float16)c[r];
  *(v8h*)(baseT + (size_t)d * ldT + sb) = h;
}

// ============================================================================
// Prep: Wt[w][n][k] = (f16) W_w[k][n], w in {q,k,v}. grid covers 3*DIM*DIM.
// ============================================================================
__global__ __launch_bounds__(256) void prep_weights_kernel(
    const float* __restrict__ Wq, const float* __restrict__ Wk,
    const float* __restrict__ Wv, _Float16* __restrict__ Wt) {
  const int per = DIM * DIM;
  int idx = blockIdx.x * 256 + threadIdx.x;  // coalesced read along n
  int w   = idx / per;
  int rem = idx - w * per;
  int k   = rem / DIM;
  int n   = rem - k * DIM;
  const float* W = (w == 0) ? Wq : (w == 1) ? Wk : Wv;
  Wt[(size_t)w * per + (size_t)n * DIM + k] = (_Float16)W[rem];
}

// ============================================================================
// Phase 1: Q/K row-major f16, V transposed f16. Block = 4 waves.
// grid = (B*S/16, DIM/64)
// ============================================================================
__global__ __launch_bounds__(128) void qkv_proj_kernel(
    const float* __restrict__ x, const _Float16* __restrict__ Wt,
    _Float16* __restrict__ Qh, _Float16* __restrict__ Kh,
    _Float16* __restrict__ Vt) {
  const int lane = threadIdx.x & 31;
  const int wave = threadIdx.x >> 5;
  const int row0 = blockIdx.x * 16;              // row in [0, B*S)
  const int col0 = blockIdx.y * 64 + wave * 16;  // col in [0, DIM)
  const int per  = DIM * DIM;

  const _Float16* Wtq = Wt;
  const _Float16* Wtk = Wt + per;
  const _Float16* Wtv = Wt + 2 * per;

  v8f cq = {}, ck = {}, cv = {};
#pragma unroll 4
  for (int k0 = 0; k0 < DIM; k0 += 32) {
    v16h a  = load_A_f32cvt(x, DIM, row0, k0, lane);
    v16h bq = load_BT_f16(Wtq, DIM, col0, k0, lane);
    cq = wmma_f16(a, bq, cq);
    v16h bk = load_BT_f16(Wtk, DIM, col0, k0, lane);
    ck = wmma_f16(a, bk, ck);
    v16h bv = load_BT_f16(Wtv, DIM, col0, k0, lane);
    cv = wmma_f16(a, bv, cv);
  }

  store_C_f16(Qh, DIM, row0, col0, lane, cq);
  store_C_f16(Kh, DIM, row0, col0, lane, ck);

  const int b    = row0 >> 12;       // SEQ = 4096 rows per batch
  const int srow = row0 & (SEQ - 1); // row tile never crosses batch boundary
  store_C_f16_T(Vt + (size_t)b * DIM * SEQ, SEQ, srow, col0, lane, cv);
}

// ============================================================================
// Phase 2: flash attention. Block = 8 waves = 256 threads, one 16-query tile.
// grid = (S/16, B)
// ============================================================================
__global__ __launch_bounds__(256) void flash_attn_kernel(
    const _Float16* __restrict__ Qh, const _Float16* __restrict__ Kh,
    const _Float16* __restrict__ Vt, float* __restrict__ out) {
  __shared__ _Float16 Qs[QT * DIM];  // 24 KB
  __shared__ float    Ss[QT * KC];   // 8 KB  raw scores
  __shared__ _Float16 Ps[QT * KC];   // 4 KB  softmax numerators
  __shared__ float m_s[QT], l_s[QT], corr_s[QT];

  const int tid  = threadIdx.x;
  const int lane = tid & 31;
  const int wave = tid >> 5;  // 0..7
  const int q0   = blockIdx.x * QT;
  const int b    = blockIdx.y;

  const _Float16* Qb  = Qh + (size_t)b * SEQ * DIM;
  const _Float16* Kb  = Kh + (size_t)b * SEQ * DIM;
  const _Float16* VtB = Vt + (size_t)b * DIM * SEQ;
  float*          Ob  = out + (size_t)b * SEQ * DIM;

  // ---- Stage Q tile (24 KB) into LDS via CDNA5 async LDS copy.
  {
    const char* src = (const char*)(Qb + (size_t)q0 * DIM);
    unsigned lds0 = (unsigned)(uintptr_t)(&Qs[0]);
#pragma unroll
    for (int i = 0; i < (QT * DIM * 2) / (256 * 16); ++i) {  // 6 x b128/thread
      unsigned lofs = lds0 + (unsigned)(tid + i * 256) * 16u;
      unsigned long long gaddr =
          (unsigned long long)(uintptr_t)src + (size_t)(tid + i * 256) * 16u;
      asm volatile("global_load_async_to_lds_b128 %0, %1, off"
                   :: "v"(lofs), "v"(gaddr) : "memory");
    }
    asm volatile("s_wait_asynccnt 0x0" ::: "memory");
  }
  if (tid < QT) { m_s[tid] = -__builtin_inff(); l_s[tid] = 0.0f; }

  // Per-wave output accumulators: 96 columns = 6 tiles of 16.
  v8f o[6];
#pragma unroll
  for (int t = 0; t < 6; ++t) o[t] = (v8f){};

  __syncthreads();

  const float scale  = 0.03608439182435161f;  // 1/sqrt(768)
  const int nchunks = (q0 + QT + KC - 1) / KC;

  for (int c = 0; c < nchunks; ++c) {
    const int j0 = c * KC;

    // ---- 1) S = Q * K^T. Wave w computes keys j0+16w .. j0+16w+15.
    {
      v8f s = {};
      const int kcol0 = j0 + wave * 16;
#pragma unroll 4
      for (int k0 = 0; k0 < DIM; k0 += 32) {
        v16h a  = load_A_f16(Qs, DIM, 0, k0, lane);
        v16h bt = load_BT_f16(Kb, DIM, kcol0, k0, lane);
        s = wmma_f16(a, bt, s);
      }
      const int n  = lane & 15;
      const int mb = (lane >> 4) << 3;
#pragma unroll
      for (int r = 0; r < 8; ++r) Ss[(mb + r) * KC + wave * 16 + n] = s[r];
    }
    __syncthreads();

    // ---- 2) Online softmax update. 16 threads per row, width-16 shuffles.
    {
      const int r    = tid >> 4;  // row 0..15
      const int sub  = tid & 15;
      const int qrow = q0 + r;
      const float m_old = m_s[r];

      float mx = m_old;
#pragma unroll
      for (int i = 0; i < KC / 16; ++i) {
        const int j = sub + i * 16;
        float v = (j0 + j <= qrow) ? Ss[r * KC + j] * scale : -__builtin_inff();
        Ss[r * KC + j] = v;
        mx = fmaxf(mx, v);
      }
#pragma unroll
      for (int m = 8; m >= 1; m >>= 1) mx = fmaxf(mx, __shfl_xor(mx, m, 16));

      float sum = 0.0f;
#pragma unroll
      for (int i = 0; i < KC / 16; ++i) {
        const int j = sub + i * 16;
        float p = __expf(Ss[r * KC + j] - mx);
        Ps[r * KC + j] = (_Float16)p;
        sum += p;
      }
#pragma unroll
      for (int m = 8; m >= 1; m >>= 1) sum += __shfl_xor(sum, m, 16);

      if (sub == 0) {
        const float corr = __expf(m_old - mx);
        corr_s[r] = corr;
        m_s[r]    = mx;
        l_s[r]    = l_s[r] * corr + sum;
      }
    }
    __syncthreads();

    // ---- 3) O = O*corr + P * V. Each wave owns 96 output columns.
    {
      const int mb = (lane >> 4) << 3;
      float crr[8];
#pragma unroll
      for (int r = 0; r < 8; ++r) crr[r] = corr_s[mb + r];
#pragma unroll
      for (int t = 0; t < 6; ++t)
#pragma unroll
        for (int r = 0; r < 8; ++r) o[t][r] *= crr[r];

#pragma unroll
      for (int kk = 0; kk < KC; kk += 32) {
        v16h a = load_A_f16(Ps, KC, 0, kk, lane);
#pragma unroll
        for (int t = 0; t < 6; ++t) {
          const int dc = wave * 96 + t * 16;
          v16h bv = load_BT_f16(VtB, SEQ, dc, j0 + kk, lane);  // contiguous
          o[t] = wmma_f16(a, bv, o[t]);
        }
      }
    }
    __syncthreads();
  }

  // ---- Epilogue: divide by running sum, store f32.
  {
    const int n  = lane & 15;
    const int mb = (lane >> 4) << 3;
#pragma unroll
    for (int r = 0; r < 8; ++r) {
      const float inv = 1.0f / l_s[mb + r];
#pragma unroll
      for (int t = 0; t < 6; ++t) {
        const int dc = wave * 96 + t * 16;
        Ob[(size_t)(q0 + mb + r) * DIM + dc + n] = o[t][r] * inv;
      }
    }
  }
}

// ============================================================================
extern "C" void kernel_launch(void* const* d_in, const int* in_sizes, int n_in,
                              void* d_out, int out_size, void* d_ws,
                              size_t ws_size, hipStream_t stream) {
  const float* x  = (const float*)d_in[0];
  const float* Wq = (const float*)d_in[1];
  const float* Wk = (const float*)d_in[2];
  const float* Wv = (const float*)d_in[3];
  float* out = (float*)d_out;

  const size_t elems = (size_t)BATCH * SEQ * DIM;  // 12,582,912
  _Float16* Qh = (_Float16*)d_ws;                  // ws: 3*elems*2 + 3.5MB
  _Float16* Kh = Qh + elems;
  _Float16* Vt = Kh + elems;
  _Float16* Wt = Vt + elems;

  prep_weights_kernel<<<(3 * DIM * DIM) / 256, 256, 0, stream>>>(Wq, Wk, Wv, Wt);

  dim3 g1((BATCH * SEQ) / 16, DIM / 64);
  qkv_proj_kernel<<<g1, 128, 0, stream>>>(x, Wt, Qh, Kh, Vt);

  dim3 g2(SEQ / QT, BATCH);
  flash_attn_kernel<<<g2, 256, 0, stream>>>(Qh, Kh, Vt, out);
}
